// TransformerMoEDecoderLayer_49443663512008
// MI455X (gfx1250) — matrix operand order repaired and checked by reading, not compiled, not run on hardware
//
#include <hip/hip_runtime.h>
#include <hip/hip_bf16.h>
#include <hip/hip_fp16.h>

// ---------------------------------------------------------------------------
// Problem dims (compile-time, from reference)
// ---------------------------------------------------------------------------
constexpr int T   = 1024;
constexpr int S   = 1024;
constexpr int Bb  = 4;
constexpr int D   = 1024;
constexpr int H   = 16;
constexpr int F   = 4096;
constexpr int E   = 8;
constexpr int HD  = D / H;        // 64
constexpr int Ntok = T * Bb;      // 4096
constexpr int CAP  = 2 * Ntok / E; // 1024

typedef __attribute__((ext_vector_type(16))) _Float16 v16h;
typedef __attribute__((ext_vector_type(8)))  float    v8f;
typedef __attribute__((ext_vector_type(8)))  _Float16 h8;
typedef __attribute__((ext_vector_type(4)))  _Float16 h4;
typedef __attribute__((ext_vector_type(4)))  unsigned u32x4;
typedef __attribute__((ext_vector_type(8)))  unsigned u32x8;

// ---------------------------------------------------------------------------
// TDM: DMA a 2D tile (32 halves wide, `rows` tall, row stride `strideElems`
// halves) from global memory into LDS at byte offset `ldsByteOff`, with HW
// padding of 4 DWORDs every 16 DWORDs -> LDS row pitch = 40 halves (80B).
// D# layout per CDNA5 ISA ch.8 (group0: 4 SGPRs, group1: 8 SGPRs).
// Issued once per workgroup (wave 0); completion tracked by TENSORcnt.
// ---------------------------------------------------------------------------
__device__ __forceinline__ void tdm_load_tile_2d(const _Float16* gsrc,
                                                 unsigned ldsByteOff,
                                                 unsigned rows,
                                                 unsigned strideElems) {
  unsigned long long ga = (unsigned long long)(size_t)gsrc;
  u32x4 g0;
  g0[0] = 1u;                                   // count=1 (valid user D#)
  g0[1] = ldsByteOff;                           // lds_addr (bytes)
  g0[2] = (unsigned)ga;                         // global_addr[31:0]
  g0[3] = (unsigned)((ga >> 32) & 0x01FFFFFFull) | (2u << 30); // addr hi | type=2
  u32x8 g1;
  g1[0] = (1u << 16)    // data_size = 2 bytes
        | (1u << 20)    // pad_enable
        | (3u << 22)    // pad_interval: 16 DWORDs (64B of data)
        | (3u << 25);   // pad_amount:   4 DWORDs (16B pad)
  g1[1] = 0xFFFF0000u;                          // tensor_dim0 = 0x7FFFFFFF (lo)
  g1[2] = 0x7FFFu | 0xFFFF0000u;                // dim0 hi | tensor_dim1 lo
  g1[3] = 0x7FFFu | (32u << 16);                // dim1 hi | tile_dim0 = 32
  g1[4] = rows & 0xFFFFu;                       // tile_dim1 (tile_dim2 = 0)
  g1[5] = strideElems;                          // tensor_dim0_stride lo32
  g1[6] = 0u;                                   // stride hi | dim1_stride lo
  g1[7] = 0u;
  asm volatile("tensor_load_to_lds %0, %1" : : "s"(g0), "s"(g1) : "memory");
}

// ---------------------------------------------------------------------------
// fp32 -> fp16 cast (vectorized)
// ---------------------------------------------------------------------------
__global__ void cast_h_k(const float* __restrict__ in, _Float16* __restrict__ out, long n) {
  long i = ((long)blockIdx.x * blockDim.x + threadIdx.x) * 4;
  long st = (long)gridDim.x * blockDim.x * 4;
  for (; i < n; i += st) {
    float4 f = *(const float4*)(in + i);
    h4 h; h[0] = (_Float16)f.x; h[1] = (_Float16)f.y; h[2] = (_Float16)f.z; h[3] = (_Float16)f.w;
    *(h4*)(out + i) = h;
  }
}

// fp32 [R][C] -> fp16 [C][R] transpose-cast, batched over gridDim.z
__global__ __launch_bounds__(256) void cast_tr_k(const float* __restrict__ in,
                                                 _Float16* __restrict__ out,
                                                 int R, int C) {
  __shared__ float tile[32][33];
  long batch = blockIdx.z;
  in  += batch * (long)R * C;
  out += batch * (long)R * C;
  int c0 = blockIdx.x * 32, r0 = blockIdx.y * 32;
  int tx = threadIdx.x & 31, ty = threadIdx.x >> 5;  // 8 rows per pass
#pragma unroll
  for (int i = 0; i < 32; i += 8)
    tile[ty + i][tx] = in[(long)(r0 + ty + i) * C + c0 + tx];
  __syncthreads();
#pragma unroll
  for (int i = 0; i < 32; i += 8)
    out[(long)(c0 + ty + i) * R + r0 + tx] = (_Float16)tile[tx][ty + i];
}

__global__ void zero4_k(uint4* __restrict__ p, long n) {
  long i = (long)blockIdx.x * blockDim.x + threadIdx.x;
  long st = (long)gridDim.x * blockDim.x;
  uint4 z; z.x = z.y = z.z = z.w = 0u;
  for (; i < n; i += st) p[i] = z;
}

// ---------------------------------------------------------------------------
// WMMA GEMM:  C[M,N] = alpha * A[M,K] * B[K,N] + bias[N]   (f16 in, f32 acc)
//   A row-major (lda), staged via double-buffered TDM DMA.
//   B: if BT element(k,n)=B[n*ldb+k] (double-buffered TDM); if !BT
//      =B[k*ldb+n] (manual transpose staging, single buffer).
//   Batched over gridDim.z; z -> (zb=z/HH, zh=z%HH) with per-axis offsets.
// ---------------------------------------------------------------------------
template <int BM, int BN, int BK, int WR, int WC, bool BT, bool RELU>
__global__ __launch_bounds__(WR* WC * 32) void gemm_f16_k(
    const _Float16* __restrict__ A, const _Float16* __restrict__ Bm,
    const float* __restrict__ bias, float* __restrict__ Cf, _Float16* __restrict__ Ch,
    int M, int N, int K, int lda, int ldb, int ldc, int HH,
    long aOffB, long aOffH, long bOffB, long bOffH, long cOffB, long cOffH,
    long biasOff, float alpha) {
  static_assert(BK == 32, "TDM descriptor assumes BK==32");
  constexpr int NT  = WR * WC * 32;
  constexpr int BKP = BK + 8;                 // row pitch 40 halves = 80B (16B pad)
  constexpr int WTM = BM / WR, WTN = BN / WC;
  constexpr int MF = WTM / 16, NF = WTN / 16;
  constexpr int NBB = BT ? 2 : 1;             // B ping-pong only on TDM path

  __shared__ __align__(16) _Float16 As[2][BM * BKP];     // double-buffered (TDM)
  __shared__ __align__(16) _Float16 Bs[NBB][BN * BKP];

  const int z  = blockIdx.z;
  const int zb = z / HH, zh = z % HH;
  A  += zb * aOffB + zh * aOffH;
  Bm += zb * bOffB + zh * bOffH;
  const long cbase = (long)zb * cOffB + (long)zh * cOffH;
  const float* biasp = bias ? bias + (long)z * biasOff : nullptr;

  const int m0 = blockIdx.y * BM, n0 = blockIdx.x * BN;
  const int tid = threadIdx.x, lane = tid & 31, wave = tid >> 5;
  const int wr = wave / WC, wc = wave % WC;
  const int lm = lane & 15, lg = lane >> 4;

  const unsigned asOff0 = (unsigned)(size_t)&As[0][0];
  const unsigned asOff1 = (unsigned)(size_t)&As[1][0];
  const unsigned bsOff0 = (unsigned)(size_t)&Bs[0][0];
  const unsigned bsOff1 = (unsigned)(size_t)&Bs[NBB - 1][0];

  v8f acc[MF][NF] = {};

  const int nIter = K / BK;
  // ---- prologue: DMA slab 0 into buffer 0
  if (wave == 0) {
    tdm_load_tile_2d(A + (long)m0 * lda, asOff0, (unsigned)BM, (unsigned)lda);
    if constexpr (BT)
      tdm_load_tile_2d(Bm + (long)n0 * ldb, bsOff0, (unsigned)BN, (unsigned)ldb);
  }

  for (int it = 0; it < nIter; ++it) {
    const int kk = it * BK;
    const int cur = it & 1, nxt = cur ^ 1;

    // ---- stage B manually when it needs a transpose (row-major [K][N])
    if constexpr (!BT) {
      constexpr int NC = BN / 8;
      for (int c = tid; c < BK * NC; c += NT) {
        int k = c / NC, nc = (c % NC) * 8;
        uint4 v = *(const uint4*)(Bm + (long)(kk + k) * ldb + n0 + nc);
        const _Float16* hv = (const _Float16*)&v;
#pragma unroll
        for (int j = 0; j < 8; j++) Bs[0][(nc + j) * BKP + k] = hv[j];
      }
    }
    // ---- TDM pipeline: issue slab it+1 into the alternate buffer, then wait
    //      until slab `it` has landed (tensor ops complete in order).
    if (wave == 0) {
      if (it + 1 < nIter) {
        tdm_load_tile_2d(A + (long)m0 * lda + kk + BK,
                         nxt ? asOff1 : asOff0, (unsigned)BM, (unsigned)lda);
        if constexpr (BT)
          tdm_load_tile_2d(Bm + (long)n0 * ldb + kk + BK,
                           nxt ? bsOff1 : bsOff0, (unsigned)BN, (unsigned)ldb);
        __builtin_amdgcn_s_wait_tensorcnt(BT ? 2 : 1);
      } else {
        __builtin_amdgcn_s_wait_tensorcnt(0);
      }
    }
    __syncthreads();

    // ---- fragments per CDNA5 WMMA VGPR layouts
    union Fr { v16h v; uint4 u[2]; };
    Fr af[MF], bf[NF];
#pragma unroll
    for (int i = 0; i < MF; i++) {
      const uint4* ap = (const uint4*)(&As[cur][0] + (wr * WTM + i * 16 + lm) * BKP);
      af[i].u[0] = ap[lg];       // K = g*8 .. g*8+7
      af[i].u[1] = ap[2 + lg];   // K = 16+g*8 .. 16+g*8+7
    }
#pragma unroll
    for (int j = 0; j < NF; j++) {
      const uint4* bp =
          (const uint4*)(&Bs[BT ? cur : 0][0] + (wc * WTN + j * 16 + lm) * BKP);
      bf[j].u[0] = bp[lg];       // K = g*8 .. g*8+7
      bf[j].u[1] = bp[2 + lg];   // K = 16+g*8 .. 16+g*8+7
    }
#pragma unroll
    for (int i = 0; i < MF; i++)
#pragma unroll
      for (int j = 0; j < NF; j++)
        acc[i][j] = __builtin_amdgcn_wmma_f32_16x16x32_f16(
            false, af[i].v, false, bf[j].v, (short)0, acc[i][j], false, false);
    __syncthreads();
  }

  // ---- epilogue: C row m = r + 8*g (per VGPR r), col n = lane%16
#pragma unroll
  for (int j = 0; j < NF; j++) {
    int gn = n0 + wc * WTN + j * 16 + lm;
    float bv = biasp ? biasp[gn] : 0.0f;
#pragma unroll
    for (int i = 0; i < MF; i++) {
      int gm0 = m0 + wr * WTM + i * 16 + lg * 8;
#pragma unroll
      for (int r = 0; r < 8; r++) {
        float v = acc[i][j][r] * alpha + bv;
        if constexpr (RELU) v = fmaxf(v, 0.0f);
        long off = cbase + (long)(gm0 + r) * ldc + gn;
        if (Cf) Cf[off] = v;
        if (Ch) Ch[off] = (_Float16)v;
      }
    }
  }
}

// ---------------------------------------------------------------------------
// In-place row softmax on f16 scores, row length 1024, one wave per row
// ---------------------------------------------------------------------------
__global__ __launch_bounds__(256) void softmax_f16_k(_Float16* __restrict__ p) {
  constexpr int C = 1024;
  int wave = threadIdx.x >> 5, lane = threadIdx.x & 31;
  long row = (long)blockIdx.x * 8 + wave;
  _Float16* r = p + row * C;
  h8 v[4];
  float mx = -3.0e38f;
#pragma unroll
  for (int i = 0; i < 4; i++) {
    v[i] = *(const h8*)(r + (lane + i * 32) * 8);
#pragma unroll
    for (int j = 0; j < 8; j++) mx = fmaxf(mx, (float)v[i][j]);
  }
#pragma unroll
  for (int o = 16; o; o >>= 1) mx = fmaxf(mx, __shfl_xor(mx, o, 32));
  float ex[4][8], s = 0.0f;
#pragma unroll
  for (int i = 0; i < 4; i++)
#pragma unroll
    for (int j = 0; j < 8; j++) { float e = __expf((float)v[i][j] - mx); ex[i][j] = e; s += e; }
#pragma unroll
  for (int o = 16; o; o >>= 1) s += __shfl_xor(s, o, 32);
  float inv = 1.0f / s;
#pragma unroll
  for (int i = 0; i < 4; i++) {
    h8 o;
#pragma unroll
    for (int j = 0; j < 8; j++) o[j] = (_Float16)(ex[i][j] * inv);
    *(h8*)(r + (lane + i * 32) * 8) = o;
  }
}

// ---------------------------------------------------------------------------
// x = LayerNorm(a + b); writes f32 and optional f16. One block per row (1024)
// ---------------------------------------------------------------------------
__global__ __launch_bounds__(256) void add_ln_k(
    const float* __restrict__ a, const float* __restrict__ b,
    const float* __restrict__ g, const float* __restrict__ be,
    float* __restrict__ of, _Float16* __restrict__ oh) {
  constexpr int C = 1024;
  int row = blockIdx.x, t = threadIdx.x;
  const float* ar = a + (long)row * C;
  const float* br = b + (long)row * C;
  float v[4], s = 0.0f, s2 = 0.0f;
#pragma unroll
  for (int i = 0; i < 4; i++) {
    int c = t + i * 256;
    float x = ar[c] + br[c];
    v[i] = x; s += x; s2 += x * x;
  }
  __shared__ float red[2][8];
#pragma unroll
  for (int o = 16; o; o >>= 1) { s += __shfl_xor(s, o, 32); s2 += __shfl_xor(s2, o, 32); }
  int wave = t >> 5, lane = t & 31;
  if (lane == 0) { red[0][wave] = s; red[1][wave] = s2; }
  __syncthreads();
  if (wave == 0) {
    float a0 = lane < 8 ? red[0][lane] : 0.0f;
    float b0 = lane < 8 ? red[1][lane] : 0.0f;
#pragma unroll
    for (int o = 4; o; o >>= 1) { a0 += __shfl_xor(a0, o, 32); b0 += __shfl_xor(b0, o, 32); }
    if (lane == 0) { red[0][0] = a0; red[1][0] = b0; }
  }
  __syncthreads();
  float mean = red[0][0] / C;
  float var  = red[1][0] / C - mean * mean;
  float inv  = rsqrtf(var + 1e-5f);
#pragma unroll
  for (int i = 0; i < 4; i++) {
    int c = t + i * 256;
    float y = (v[i] - mean) * inv * g[c] + be[c];
    if (of) of[(long)row * C + c] = y;
    if (oh) oh[(long)row * C + c] = (_Float16)y;
  }
}

// ---------------------------------------------------------------------------
// MoE gating: logits = x @ wg [N,8], softmax, top1/top2, renormalized weights.
// One block per token, wave e computes logit e.
// ---------------------------------------------------------------------------
__global__ __launch_bounds__(256) void gate_k(
    const float* __restrict__ x, const float* __restrict__ wg,
    int* __restrict__ i1, int* __restrict__ i2,
    float* __restrict__ g1, float* __restrict__ g2) {
  int n = blockIdx.x, t = threadIdx.x, wave = t >> 5, lane = t & 31;
  const float* xr = x + (long)n * D;
  float s = 0.0f;
  for (int d = lane; d < D; d += 32) s += xr[d] * wg[(long)d * E + wave];
#pragma unroll
  for (int o = 16; o; o >>= 1) s += __shfl_xor(s, o, 32);
  __shared__ float lg[E];
  if (lane == 0) lg[wave] = s;
  __syncthreads();
  if (t == 0) {
    float mx = lg[0];
    for (int e = 1; e < E; e++) mx = fmaxf(mx, lg[e]);
    float ex[E], sum = 0.0f;
    for (int e = 0; e < E; e++) { ex[e] = __expf(lg[e] - mx); sum += ex[e]; }
    int a1 = 0;
    for (int e = 1; e < E; e++) if (lg[e] > lg[a1]) a1 = e;
    int a2 = -1;
    for (int e = 0; e < E; e++) {
      if (e == a1) continue;
      if (a2 < 0 || lg[e] > lg[a2]) a2 = e;
    }
    float p1 = ex[a1] / sum, p2 = ex[a2] / sum;
    float den = p1 + p2 + 1e-9f;
    i1[n] = a1; i2[n] = a2; g1[n] = p1 / den; g2[n] = p2 / den;
  }
}

// ---------------------------------------------------------------------------
// Capacity positions: wave32 ballot prefix-sum, one wave per expert,
// token order preserved (matches jnp.cumsum FCFS semantics). 1 block.
// ---------------------------------------------------------------------------
__global__ __launch_bounds__(256) void positions_k(
    const int* __restrict__ i1, const int* __restrict__ i2,
    int* __restrict__ p1, int* __restrict__ p2) {
  __shared__ int counts1[E];
  int wave = threadIdx.x >> 5, lane = threadIdx.x & 31;
  int e = wave, cnt = 0;
  for (int base = 0; base < Ntok; base += 32) {
    int t = base + lane;
    bool pr = (i1[t] == e);
    unsigned mm = (unsigned)__ballot(pr);
    if (pr) p1[t] = cnt + __popc(mm & ((1u << lane) - 1u));
    cnt += __popc(mm);
  }
  if (lane == 0) counts1[e] = cnt;
  __syncthreads();
  int b0 = counts1[e], cnt2 = 0;
  for (int base = 0; base < Ntok; base += 32) {
    int t = base + lane;
    bool pr = (i2[t] == e);
    unsigned mm = (unsigned)__ballot(pr);
    if (pr) p2[t] = b0 + cnt2 + __popc(mm & ((1u << lane) - 1u));
    cnt2 += __popc(mm);
  }
}

// ---------------------------------------------------------------------------
// Scatter token rows (f16) into expert capacity buffers
// ---------------------------------------------------------------------------
__global__ __launch_bounds__(256) void scatter_k(
    const _Float16* __restrict__ x16, const int* __restrict__ i1,
    const int* __restrict__ p1, const int* __restrict__ i2,
    const int* __restrict__ p2, _Float16* __restrict__ buf) {
  int n = blockIdx.x, t = threadIdx.x;
  uint2 v = ((const uint2*)(x16 + (long)n * D))[t];  // 4 halves per thread
  int q1 = p1[n], q2 = p2[n];
  if (q1 < CAP) ((uint2*)(buf + ((long)i1[n] * CAP + q1) * D))[t] = v;
  if (q2 < CAP) ((uint2*)(buf + ((long)i2[n] * CAP + q2) * D))[t] = v;
}

// ---------------------------------------------------------------------------
// Gather expert outputs, combine, residual + final LayerNorm -> d_out (f32)
// ---------------------------------------------------------------------------
__global__ __launch_bounds__(256) void combine_ln_k(
    const float* __restrict__ x2, const float* __restrict__ eo,
    const int* __restrict__ i1, const int* __restrict__ p1,
    const int* __restrict__ i2, const int* __restrict__ p2,
    const float* __restrict__ g1, const float* __restrict__ g2,
    const float* __restrict__ g, const float* __restrict__ be,
    float* __restrict__ out) {
  constexpr int C = 1024;
  int n = blockIdx.x, t = threadIdx.x;
  int q1 = p1[n], q2 = p2[n];
  float k1 = (q1 < CAP) ? g1[n] : 0.0f;
  float k2 = (q2 < CAP) ? g2[n] : 0.0f;
  long o1 = ((long)i1[n] * CAP + (q1 < CAP ? q1 : CAP - 1)) * D;
  long o2 = ((long)i2[n] * CAP + (q2 < CAP ? q2 : CAP - 1)) * D;
  const float* xr = x2 + (long)n * C;
  float v[4], s = 0.0f, s2 = 0.0f;
#pragma unroll
  for (int i = 0; i < 4; i++) {
    int c = t + i * 256;
    float m = eo[o1 + c] * k1 + eo[o2 + c] * k2;
    float x = xr[c] + m;
    v[i] = x; s += x; s2 += x * x;
  }
  __shared__ float red[2][8];
#pragma unroll
  for (int o = 16; o; o >>= 1) { s += __shfl_xor(s, o, 32); s2 += __shfl_xor(s2, o, 32); }
  int wave = t >> 5, lane = t & 31;
  if (lane == 0) { red[0][wave] = s; red[1][wave] = s2; }
  __syncthreads();
  if (wave == 0) {
    float a0 = lane < 8 ? red[0][lane] : 0.0f;
    float b0 = lane < 8 ? red[1][lane] : 0.0f;
#pragma unroll
    for (int o = 4; o; o >>= 1) { a0 += __shfl_xor(a0, o, 32); b0 += __shfl_xor(b0, o, 32); }
    if (lane == 0) { red[0][0] = a0; red[1][0] = b0; }
  }
  __syncthreads();
  float mean = red[0][0] / C;
  float var  = red[1][0] / C - mean * mean;
  float inv  = rsqrtf(var + 1e-5f);
#pragma unroll
  for (int i = 0; i < 4; i++) {
    int c = t + i * 256;
    out[(long)n * C + c] = (v[i] - mean) * inv * g[c] + be[c];
  }
}

// ---------------------------------------------------------------------------
// Host orchestration
// ---------------------------------------------------------------------------
extern "C" void kernel_launch(void* const* d_in, const int* in_sizes, int n_in,
                              void* d_out, int out_size, void* d_ws, size_t ws_size,
                              hipStream_t stream) {
  const float* tgt      = (const float*)d_in[0];
  const float* memry    = (const float*)d_in[1];
  const float* sa_in_w  = (const float*)d_in[2];
  const float* sa_in_b  = (const float*)d_in[3];
  const float* sa_out_w = (const float*)d_in[4];
  const float* sa_out_b = (const float*)d_in[5];
  const float* ca_in_w  = (const float*)d_in[6];
  const float* ca_in_b  = (const float*)d_in[7];
  const float* ca_out_w = (const float*)d_in[8];
  const float* ca_out_b = (const float*)d_in[9];
  const float* ln1_g = (const float*)d_in[10];
  const float* ln1_b = (const float*)d_in[11];
  const float* ln2_g = (const float*)d_in[12];
  const float* ln2_b = (const float*)d_in[13];
  const float* ln3_g = (const float*)d_in[14];
  const float* ln3_b = (const float*)d_in[15];
  const float* wg = (const float*)d_in[16];
  const float* w1 = (const float*)d_in[17];
  const float* b1 = (const float*)d_in[18];
  const float* w2 = (const float*)d_in[19];
  const float* b2 = (const float*)d_in[20];

  char* wsp = (char*)d_ws;
  auto alloc = [&](size_t bytes) -> char* {
    char* p = wsp;
    wsp += (bytes + 255) & ~(size_t)255;
    return p;
  };
  _Float16* tgt16   = (_Float16*)alloc((size_t)Ntok * D * 2);
  _Float16* mem16   = (_Float16*)alloc((size_t)Ntok * D * 2);
  _Float16* saw16   = (_Float16*)alloc((size_t)3 * D * D * 2);
  _Float16* saow16  = (_Float16*)alloc((size_t)D * D * 2);
  _Float16* caw16   = (_Float16*)alloc((size_t)3 * D * D * 2);
  _Float16* caow16  = (_Float16*)alloc((size_t)D * D * 2);
  _Float16* w1T16   = (_Float16*)alloc((size_t)E * D * F * 2);  // [E][F][D]
  _Float16* w2T16   = (_Float16*)alloc((size_t)E * F * D * 2);  // [E][D][F]
  _Float16* qkv16   = (_Float16*)alloc((size_t)Ntok * 3 * D * 2);
  _Float16* scores16= (_Float16*)alloc((size_t)Bb * H * T * S * 2);
  _Float16* ao16    = (_Float16*)alloc((size_t)Ntok * D * 2);
  float*    t2      = (float*)   alloc((size_t)Ntok * D * 4);
  float*    xf      = (float*)   alloc((size_t)Ntok * D * 4);
  _Float16* x16     = (_Float16*)alloc((size_t)Ntok * D * 2);
  float*    x2f     = (float*)   alloc((size_t)Ntok * D * 4);
  _Float16* x216    = (_Float16*)alloc((size_t)Ntok * D * 2);
  _Float16* qca16   = (_Float16*)alloc((size_t)Ntok * D * 2);
  _Float16* kvca16  = (_Float16*)alloc((size_t)Ntok * 2 * D * 2);
  int*      idx1    = (int*)     alloc((size_t)Ntok * 4);
  int*      idx2    = (int*)     alloc((size_t)Ntok * 4);
  int*      pos1    = (int*)     alloc((size_t)Ntok * 4);
  int*      pos2    = (int*)     alloc((size_t)Ntok * 4);
  float*    gg1     = (float*)   alloc((size_t)Ntok * 4);
  float*    gg2     = (float*)   alloc((size_t)Ntok * 4);
  _Float16* buf16   = (_Float16*)alloc((size_t)E * CAP * D * 2);
  _Float16* h16     = (_Float16*)alloc((size_t)E * CAP * F * 2);
  float*    eo      = (float*)   alloc((size_t)E * CAP * D * 4);

  // ---- casts to f16 (MoE weights transpose-cast so their GEMMs are TDM-stageable)
  auto cast = [&](const float* src, _Float16* dst, long n) {
    cast_h_k<<<4096, 256, 0, stream>>>(src, dst, n);
  };
  cast(tgt, tgt16, (long)Ntok * D);
  cast(memry, mem16, (long)Ntok * D);
  cast(sa_in_w, saw16, (long)3 * D * D);
  cast(sa_out_w, saow16, (long)D * D);
  cast(ca_in_w, caw16, (long)3 * D * D);
  cast(ca_out_w, caow16, (long)D * D);
  cast_tr_k<<<dim3(F / 32, D / 32, E), 256, 0, stream>>>(w1, w1T16, D, F);
  cast_tr_k<<<dim3(D / 32, F / 32, E), 256, 0, stream>>>(w2, w2T16, F, D);

  // ---- 1) SA QKV: qkv16[N,3D] = tgt16 @ sa_in_w^T + b   (B transposed: W[o,k])
  gemm_f16_k<128, 128, 32, 2, 4, true, false>
      <<<dim3(3 * D / 128, Ntok / 128, 1), 256, 0, stream>>>(
          tgt16, saw16, sa_in_b, nullptr, qkv16,
          Ntok, 3 * D, D, D, D, 3 * D, 1, 0, 0, 0, 0, 0, 0, 0, 1.0f);

  // ---- 2) SA scores[b,h,t,s] = q . k / 8   (batched over 64 (b,h))
  gemm_f16_k<128, 128, 32, 2, 4, true, false>
      <<<dim3(S / 128, T / 128, Bb * H), 256, 0, stream>>>(
          qkv16, qkv16 + D, nullptr, nullptr, scores16,
          T, S, HD, Bb * 3 * D, Bb * 3 * D, S, H,
          (long)3 * D, (long)HD, (long)3 * D, (long)HD,
          (long)H * T * S, (long)T * S, 0, 0.125f);

  // ---- 3) softmax in-place
  softmax_f16_k<<<Bb * H * T / 8, 256, 0, stream>>>(scores16);

  // ---- 4) attn @ V -> ao16[t,b,h*64+d]
  gemm_f16_k<128, 64, 32, 4, 2, false, false>
      <<<dim3(1, T / 128, Bb * H), 256, 0, stream>>>(
          scores16, qkv16 + 2 * D, nullptr, nullptr, ao16,
          T, HD, S, S, Bb * 3 * D, Bb * D, H,
          (long)H * T * S, (long)T * S, (long)3 * D, (long)HD,
          (long)D, (long)HD, 0, 1.0f);

  // ---- 5) SA out proj -> t2 (f32)
  gemm_f16_k<128, 128, 32, 2, 4, true, false>
      <<<dim3(D / 128, Ntok / 128, 1), 256, 0, stream>>>(
          ao16, saow16, sa_out_b, t2, nullptr,
          Ntok, D, D, D, D, D, 1, 0, 0, 0, 0, 0, 0, 0, 1.0f);

  // ---- 6) x = LN(tgt + t2)
  add_ln_k<<<Ntok, 256, 0, stream>>>(tgt, t2, ln1_g, ln1_b, xf, x16);

  // ---- 7) CA q proj
  gemm_f16_k<128, 128, 32, 2, 4, true, false>
      <<<dim3(D / 128, Ntok / 128, 1), 256, 0, stream>>>(
          x16, caw16, ca_in_b, nullptr, qca16,
          Ntok, D, D, D, D, D, 1, 0, 0, 0, 0, 0, 0, 0, 1.0f);

  // ---- 8) CA kv proj (memory)
  gemm_f16_k<128, 128, 32, 2, 4, true, false>
      <<<dim3(2 * D / 128, Ntok / 128, 1), 256, 0, stream>>>(
          mem16, caw16 + (long)D * D, ca_in_b + D, nullptr, kvca16,
          Ntok, 2 * D, D, D, D, 2 * D, 1, 0, 0, 0, 0, 0, 0, 0, 1.0f);

  // ---- 9) CA scores
  gemm_f16_k<128, 128, 32, 2, 4, true, false>
      <<<dim3(S / 128, T / 128, Bb * H), 256, 0, stream>>>(
          qca16, kvca16, nullptr, nullptr, scores16,
          T, S, HD, Bb * D, Bb * 2 * D, S, H,
          (long)D, (long)HD, (long)2 * D, (long)HD,
          (long)H * T * S, (long)T * S, 0, 0.125f);

  // ---- 10) softmax
  softmax_f16_k<<<Bb * H * T / 8, 256, 0, stream>>>(scores16);

  // ---- 11) CA attn @ V
  gemm_f16_k<128, 64, 32, 4, 2, false, false>
      <<<dim3(1, T / 128, Bb * H), 256, 0, stream>>>(
          scores16, kvca16 + D, nullptr, nullptr, ao16,
          T, HD, S, S, Bb * 2 * D, Bb * D, H,
          (long)H * T * S, (long)T * S, (long)2 * D, (long)HD,
          (long)D, (long)HD, 0, 1.0f);

  // ---- 12) CA out proj -> t2
  gemm_f16_k<128, 128, 32, 2, 4, true, false>
      <<<dim3(D / 128, Ntok / 128, 1), 256, 0, stream>>>(
          ao16, caow16, ca_out_b, t2, nullptr,
          Ntok, D, D, D, D, D, 1, 0, 0, 0, 0, 0, 0, 0, 1.0f);

  // ---- 13) x2 = LN(x + t2)
  add_ln_k<<<Ntok, 256, 0, stream>>>(xf, t2, ln2_g, ln2_b, x2f, x216);

  // ---- 14-17) MoE routing
  gate_k<<<Ntok, 256, 0, stream>>>(x2f, wg, idx1, idx2, gg1, gg2);
  positions_k<<<1, 256, 0, stream>>>(idx1, idx2, pos1, pos2);
  zero4_k<<<2048, 256, 0, stream>>>((uint4*)buf16, (long)E * CAP * D / 8);
  scatter_k<<<Ntok, 256, 0, stream>>>(x216, idx1, pos1, idx2, pos2, buf16);

  // ---- 18) expert FFN layer 1: h = relu(buf @ w1 + b1)  (w1T: [E][F][D], BT)
  gemm_f16_k<128, 128, 32, 2, 4, true, true>
      <<<dim3(F / 128, CAP / 128, E), 256, 0, stream>>>(
          buf16, w1T16, b1, nullptr, h16,
          CAP, F, D, D, D, F, 1,
          (long)CAP * D, 0, (long)F * D, 0, (long)CAP * F, 0, (long)F, 1.0f);

  // ---- 19) expert FFN layer 2: eo = h @ w2 + b2  (w2T: [E][D][F], BT)
  gemm_f16_k<128, 128, 32, 2, 4, true, false>
      <<<dim3(D / 128, CAP / 128, E), 256, 0, stream>>>(
          h16, w2T16, b2, eo, nullptr,
          CAP, D, F, F, F, D, 1,
          (long)CAP * F, 0, (long)D * F, 0, (long)CAP * D, 0, (long)D, 1.0f);

  // ---- 20) gather + combine + residual + LN3 -> out
  combine_ln_k<<<Ntok, 256, 0, stream>>>(x2f, eo, idx1, pos1, idx2, pos2,
                                         gg1, gg2, ln3_g, ln3_b, (float*)d_out);
}
